// TransformerLiteFusion_60052232732995
// MI455X (gfx1250) — matrix operand order repaired and checked by reading, not compiled, not run on hardware
//
#include <hip/hip_runtime.h>
#include <hip/hip_bf16.h>
#include <cstdint>
#include <cstddef>

// ---- CDNA5 (gfx1250) types ----
typedef __attribute__((ext_vector_type(16))) __bf16 v16bf;
typedef __attribute__((ext_vector_type(8)))  float  v8f;
typedef __attribute__((ext_vector_type(4)))  float  v4f;

#define WAVES_PER_BLOCK  4
#define THREADS_PER_BLOCK (WAVES_PER_BLOCK * 32)
#define SAMPLES_PER_WAVE 4   // 4 samples x M=4 tokens = 16 WMMA rows per wave

// CDNA5 async global->LDS copy via inline asm (ASYNCcnt-tracked).
#define TLF_ASYNC_LDS 1

// A-matrix 16x32 bf16 fragment (ISA 7.12.2): lane = (M row) | hi selects K half.
// VGPR0..3: K=0..7 (+8 if hi); VGPR4..7: K=16..23 (+8 if hi).
// Uses hardware f32->bf16 converts (v_cvt_*bf16*) instead of bit tricks.
__device__ __forceinline__ v16bf make_afrag(const float* rowbase, int hi) {
  v16bf a;
#pragma unroll
  for (int i = 0; i < 16; ++i) {
    int j = i >> 1;
    int k = ((j < 4) ? (2 * j) : (16 + 2 * (j - 4))) + (i & 1) + (hi ? 8 : 0);
    a[i] = (__bf16)rowbase[k];
  }
  return a;
}

// cross-lane LDS phase fence (compiler cannot see cross-lane LDS deps)
#define LDS_FENCE() asm volatile("s_wait_dscnt 0x0" ::: "memory")

__global__ __launch_bounds__(THREADS_PER_BLOCK)
void tlf_fused_kernel(const float* __restrict__ tokens,
                      const float* __restrict__ amask,
                      const float* __restrict__ type_emb,
                      const float* __restrict__ in_proj_w,
                      const float* __restrict__ in_proj_b,
                      const float* __restrict__ out_proj_w,
                      const float* __restrict__ out_proj_b,
                      const float* __restrict__ ln1_s,
                      const float* __restrict__ ln1_b,
                      const float* __restrict__ w1p,
                      const float* __restrict__ b1p,
                      const float* __restrict__ w2p,
                      const float* __restrict__ b2p,
                      const float* __restrict__ ln2_s,
                      const float* __restrict__ ln2_b,
                      const float* __restrict__ gate_w,
                      const float* __restrict__ gate_b,
                      float* __restrict__ out_fused,
                      float* __restrict__ out_weights,
                      int Ntotal) {
  // B fragments (bf16) for 12 tiles: qkv[0..5], out[6..7], w1[8..9], w2[10..11]
  __shared__ __align__(32) __bf16 sB[12 * 32 * 16];             // 12 KB
  // constants: [0..95]=in_b, [96..127]=out_b, [128..159]=b1, [160..191]=b2,
  // [192..223]=ln1s,[224..255]=ln1b,[256..287]=ln2s,[288..319]=ln2b,
  // [320..447]=type_emb, [448..479]=gate_w, [480]=gate_b
  __shared__ float sC[481];
  __shared__ __align__(16) float sX[WAVES_PER_BLOCK][16 * 32];  // x / ln outputs
  __shared__ __align__(16) float sQ[WAVES_PER_BLOCK][16 * 96];  // qkv
  __shared__ __align__(16) float sT[WAVES_PER_BLOCK][16 * 32];  // ctx / pre-LN / relu(h)

  const int tid  = threadIdx.x;
  const int wv   = tid >> 5;
  const int lane = tid & 31;

  // ---- stage weights into WMMA B-fragment layout (bf16) ----
  // B 32x16 bf16 (ISA 7.12.2): VGPR j lanes0-15 K={2j,2j+1} (+8 if j>=4); lanes16-31 K+=8.
  // => lane l, element i: K = (l>=16?8:0) + i + (i>=8?8:0), N = l&15. B[k][n] = W[fbase+n][k].
  for (int idx = tid; idx < 12 * 32; idx += THREADS_PER_BLOCK) {
    int tile = idx >> 5, l = idx & 31;
    int n = l & 15, khalf = (l >> 4) << 3;
    const float* Wsrc; int fbase;
    if (tile < 6)       { Wsrc = in_proj_w;  fbase = tile * 16; }
    else if (tile < 8)  { Wsrc = out_proj_w; fbase = (tile - 6) * 16; }
    else if (tile < 10) { Wsrc = w1p;        fbase = (tile - 8) * 16; }
    else                { Wsrc = w2p;        fbase = (tile - 10) * 16; }
    const float* col = Wsrc + (size_t)(fbase + n) * 32;
    __bf16* dst = &sB[idx * 16];
#pragma unroll
    for (int i = 0; i < 16; ++i) dst[i] = (__bf16)col[khalf + i + (i >= 8 ? 8 : 0)];
  }
  for (int i = tid; i < 96; i += THREADS_PER_BLOCK) sC[i] = in_proj_b[i];
  for (int i = tid; i < 32; i += THREADS_PER_BLOCK) {
    sC[96 + i]  = out_proj_b[i]; sC[128 + i] = b1p[i];  sC[160 + i] = b2p[i];
    sC[192 + i] = ln1_s[i];      sC[224 + i] = ln1_b[i];
    sC[256 + i] = ln2_s[i];      sC[288 + i] = ln2_b[i];
    sC[448 + i] = gate_w[i];
  }
  for (int i = tid; i < 128; i += THREADS_PER_BLOCK) sC[320 + i] = type_emb[i];
  if (tid == 0) sC[480] = gate_b[0];
  __syncthreads();

  const int sampleBase = (blockIdx.x * WAVES_PER_BLOCK + wv) * SAMPLES_PER_WAVE;
  if (sampleBase >= Ntotal) return;   // uniform per wave; no barriers below

  float* myX = sX[wv];
  float* myQ = sQ[wv];
  float* myT = sT[wv];

  // ---- stream 16 rows (4 samples x 4 tokens x 32 f32) into LDS ----
  const float* gsrc = tokens + (size_t)sampleBase * 128;
#ifdef TLF_ASYNC_LDS
#pragma unroll
  for (int c = 0; c < 4; ++c) {
    // VDST = LDS byte address (low 32 bits of flat pointer to __shared__),
    // VADDR = 64-bit global address, GV mode (saddr = off).
    unsigned lds_addr = (unsigned)(uintptr_t)(myX + c * 128 + lane * 4);
    unsigned long long ga = (unsigned long long)(uintptr_t)(gsrc + c * 128 + lane * 4);
    asm volatile("global_load_async_to_lds_b128 %0, %1, off"
                 :: "v"(lds_addr), "v"(ga)
                 : "memory");
  }
  asm volatile("s_wait_asynccnt 0x0" ::: "memory");
#else
#pragma unroll
  for (int c = 0; c < 4; ++c)
    *(v4f*)(myX + c * 128 + lane * 4) = *(const v4f*)(gsrc + c * 128 + lane * 4);
#endif
  LDS_FENCE();
  // x = tokens + type_emb   (te index == linear index mod 128)
#pragma unroll
  for (int c = 0; c < 4; ++c) {
    int e = c * 128 + lane * 4;
    v4f t = *(v4f*)(myX + e);
#pragma unroll
    for (int j = 0; j < 4; ++j) t[j] += sC[320 + ((e + j) & 127)];
    *(v4f*)(myX + e) = t;
  }
  LDS_FENCE();

  const int mr  = lane & 15;       // WMMA row (M) for A / output col (N) for C
  const int hi  = lane >> 4;       // K-half select for A frags
  const int hi8 = hi << 3;         // C/D row offset (M=j or j+8)
  const int nc  = lane & 15;

  // ---- QKV: [16x32] x [32x96] = 6 WMMAs ----
  {
    v16bf a = make_afrag(myX + mr * 32, hi);
    v8f qk[6];
#pragma unroll
    for (int t = 0; t < 6; ++t) {
      v16bf b = *reinterpret_cast<const v16bf*>(&sB[(t * 32 + lane) * 16]);
      v8f c = {0.f, 0.f, 0.f, 0.f, 0.f, 0.f, 0.f, 0.f};
      qk[t] = __builtin_amdgcn_wmma_f32_16x16x32_bf16(false, a, false, b,
                                                      (short)0, c, false, false);
    }
#pragma unroll
    for (int t = 0; t < 6; ++t)
#pragma unroll
      for (int j = 0; j < 8; ++j) {
        int row = hi8 + j, col = t * 16 + nc;
        myQ[row * 96 + col] = qk[t][j] + sC[col];
      }
  }
  LDS_FENCE();

  // ---- attention: one lane per (sample, head, row-pair); M=4, HD=8 ----
  {
    const int sl = lane >> 3, hh = (lane >> 1) & 3, rp = lane & 1;
    const float inv_sqrt_hd = 0.35355339059327373f;
    float kbias[4];
#pragma unroll
    for (int km = 0; km < 4; ++km) {
      float av = amask[(size_t)(sampleBase + sl) * 4 + km];
      kbias[km] = (av <= 0.f) ? -1e9f : 0.f;
    }
    // k/v rows for this (sample, head): 2x v4f each -> ds_load_b128
    v4f kr[4][2], vr[4][2];
#pragma unroll
    for (int km = 0; km < 4; ++km) {
      const float* kp = myQ + (sl * 4 + km) * 96 + 32 + hh * 8;
      const float* vp = myQ + (sl * 4 + km) * 96 + 64 + hh * 8;
      kr[km][0] = *(const v4f*)kp;     kr[km][1] = *(const v4f*)(kp + 4);
      vr[km][0] = *(const v4f*)vp;     vr[km][1] = *(const v4f*)(vp + 4);
    }
#pragma unroll
    for (int mi = 0; mi < 2; ++mi) {
      int m = rp * 2 + mi;
      const float* qp = myQ + (sl * 4 + m) * 96 + hh * 8;
      v4f q0 = *(const v4f*)qp, q1 = *(const v4f*)(qp + 4);
      float sc[4], mx = -3.0e38f;
#pragma unroll
      for (int km = 0; km < 4; ++km) {
        float d = 0.f;
#pragma unroll
        for (int e2 = 0; e2 < 4; ++e2) d += q0[e2] * kr[km][0][e2] + q1[e2] * kr[km][1][e2];
        sc[km] = d * inv_sqrt_hd + kbias[km];
        mx = fmaxf(mx, sc[km]);
      }
      float se = 0.f, ex[4];
#pragma unroll
      for (int km = 0; km < 4; ++km) { ex[km] = __expf(sc[km] - mx); se += ex[km]; }
      float isum = 1.f / se;
      v4f c0 = {0.f, 0.f, 0.f, 0.f}, c1 = {0.f, 0.f, 0.f, 0.f};
#pragma unroll
      for (int km = 0; km < 4; ++km) {
        float wk = ex[km] * isum;
#pragma unroll
        for (int e2 = 0; e2 < 4; ++e2) { c0[e2] += wk * vr[km][0][e2]; c1[e2] += wk * vr[km][1][e2]; }
      }
      float* crow = myT + (sl * 4 + m) * 32 + hh * 8;
      *(v4f*)crow = c0;
      *(v4f*)(crow + 4) = c1;
    }
  }
  LDS_FENCE();

  // ---- out_proj + residual (2 WMMAs), pre-LN1 into myT ----
  {
    v16bf a = make_afrag(myT + mr * 32, hi);
    v8f o[2];
#pragma unroll
    for (int t = 0; t < 2; ++t) {
      v16bf b = *reinterpret_cast<const v16bf*>(&sB[((6 + t) * 32 + lane) * 16]);
      v8f c = {0.f, 0.f, 0.f, 0.f, 0.f, 0.f, 0.f, 0.f};
      o[t] = __builtin_amdgcn_wmma_f32_16x16x32_bf16(false, a, false, b,
                                                     (short)0, c, false, false);
    }
    LDS_FENCE();  // all ctx reads done before overwriting myT
#pragma unroll
    for (int t = 0; t < 2; ++t)
#pragma unroll
      for (int j = 0; j < 8; ++j) {
        int row = hi8 + j, col = t * 16 + nc;
        myT[row * 32 + col] = myX[row * 32 + col] + o[t][j] + sC[96 + col];
      }
  }
  LDS_FENCE();

  // ---- LN1: myT -> myX (two lanes per row, shfl_xor(16) combine) ----
  {
    const int r = lane & 15, hf = lane >> 4;
    const float* row = myT + r * 32 + hf * 16;
    v4f rv[4];
    float s = 0.f, s2 = 0.f;
#pragma unroll
    for (int q = 0; q < 4; ++q) {
      rv[q] = *(const v4f*)(row + q * 4);
#pragma unroll
      for (int i = 0; i < 4; ++i) { float v = rv[q][i]; s += v; s2 += v * v; }
    }
    s  += __shfl_xor(s, 16, 32);
    s2 += __shfl_xor(s2, 16, 32);
    float mu = s * (1.f / 32.f);
    float rs = rsqrtf(s2 * (1.f / 32.f) - mu * mu + 1e-5f);
    float* drow = myX + r * 32 + hf * 16;
#pragma unroll
    for (int q = 0; q < 4; ++q) {
      v4f o;
#pragma unroll
      for (int i = 0; i < 4; ++i) {
        int e2 = hf * 16 + q * 4 + i;
        o[i] = (rv[q][i] - mu) * rs * sC[192 + e2] + sC[224 + e2];
      }
      *(v4f*)(drow + q * 4) = o;
    }
  }
  LDS_FENCE();

  // ---- FFN layer 1 (2 WMMAs) -> relu -> myT ----
  {
    v16bf a = make_afrag(myX + mr * 32, hi);
    v8f h[2];
#pragma unroll
    for (int t = 0; t < 2; ++t) {
      v16bf b = *reinterpret_cast<const v16bf*>(&sB[((8 + t) * 32 + lane) * 16]);
      v8f c = {0.f, 0.f, 0.f, 0.f, 0.f, 0.f, 0.f, 0.f};
      h[t] = __builtin_amdgcn_wmma_f32_16x16x32_bf16(false, a, false, b,
                                                     (short)0, c, false, false);
    }
    LDS_FENCE();
#pragma unroll
    for (int t = 0; t < 2; ++t)
#pragma unroll
      for (int j = 0; j < 8; ++j) {
        int row = hi8 + j, col = t * 16 + nc;
        myT[row * 32 + col] = fmaxf(h[t][j] + sC[128 + col], 0.f);
      }
  }
  LDS_FENCE();

  // ---- FFN layer 2 (2 WMMAs) + residual, pre-LN2 into myT ----
  {
    v16bf a = make_afrag(myT + mr * 32, hi);
    v8f f2[2];
#pragma unroll
    for (int t = 0; t < 2; ++t) {
      v16bf b = *reinterpret_cast<const v16bf*>(&sB[((10 + t) * 32 + lane) * 16]);
      v8f c = {0.f, 0.f, 0.f, 0.f, 0.f, 0.f, 0.f, 0.f};
      f2[t] = __builtin_amdgcn_wmma_f32_16x16x32_bf16(false, a, false, b,
                                                      (short)0, c, false, false);
    }
    LDS_FENCE();
#pragma unroll
    for (int t = 0; t < 2; ++t)
#pragma unroll
      for (int j = 0; j < 8; ++j) {
        int row = hi8 + j, col = t * 16 + nc;
        myT[row * 32 + col] = myX[row * 32 + col] + f2[t][j] + sC[160 + col];
      }
  }
  LDS_FENCE();

  // ---- LN2: myT -> myX ----
  {
    const int r = lane & 15, hf = lane >> 4;
    const float* row = myT + r * 32 + hf * 16;
    v4f rv[4];
    float s = 0.f, s2 = 0.f;
#pragma unroll
    for (int q = 0; q < 4; ++q) {
      rv[q] = *(const v4f*)(row + q * 4);
#pragma unroll
      for (int i = 0; i < 4; ++i) { float v = rv[q][i]; s += v; s2 += v * v; }
    }
    s  += __shfl_xor(s, 16, 32);
    s2 += __shfl_xor(s2, 16, 32);
    float mu = s * (1.f / 32.f);
    float rs = rsqrtf(s2 * (1.f / 32.f) - mu * mu + 1e-5f);
    float* drow = myX + r * 32 + hf * 16;
#pragma unroll
    for (int q = 0; q < 4; ++q) {
      v4f o;
#pragma unroll
      for (int i = 0; i < 4; ++i) {
        int e2 = hf * 16 + q * 4 + i;
        o[i] = (rv[q][i] - mu) * rs * sC[256 + e2] + sC[288 + e2];
      }
      *(v4f*)(drow + q * 4) = o;
    }
  }
  LDS_FENCE();

  // ---- gate + masked-softmax pooling ----
  float wts[4];
  {
    const int r = lane & 15, hf = lane >> 4;
    const float* xr = myX + r * 32 + hf * 16;
    float g = 0.f;
#pragma unroll
    for (int q = 0; q < 4; ++q) {
      v4f xv = *(const v4f*)(xr + q * 4);
#pragma unroll
      for (int i = 0; i < 4; ++i) g += xv[i] * sC[448 + hf * 16 + q * 4 + i];
    }
    g += __shfl_xor(g, 16, 32);
    g += sC[480];
    const int smp = r >> 2, mrow = r & 3;
    const float* mrowp = amask + (size_t)(sampleBase + smp) * 4;
    float av_self = mrowp[mrow];
    float lg = (av_self <= 0.f) ? -1e9f : g;
    int base = (lane & 16) | (r & 12);   // gather within my half-wave
    float l0 = __shfl(lg, base + 0, 32);
    float l1 = __shfl(lg, base + 1, 32);
    float l2 = __shfl(lg, base + 2, 32);
    float l3 = __shfl(lg, base + 3, 32);
    float mx = fmaxf(fmaxf(l0, l1), fmaxf(l2, l3));
    float e0 = __expf(l0 - mx), e1 = __expf(l1 - mx);
    float e2 = __expf(l2 - mx), e3 = __expf(l3 - mx);
    float inv = 1.f / (e0 + e1 + e2 + e3);
    float p0 = e0 * inv * mrowp[0], p1 = e1 * inv * mrowp[1];
    float p2 = e2 * inv * mrowp[2], p3 = e3 * inv * mrowp[3];
    float ip = 1.f / (p0 + p1 + p2 + p3 + 1e-8f);
    wts[0] = p0 * ip; wts[1] = p1 * ip; wts[2] = p2 * ip; wts[3] = p3 * ip;
    if (hf == 0)
      out_weights[(size_t)(sampleBase + smp) * 4 + mrow] = wts[mrow];
  }

  // ---- fused = sum_m x[m] * w[m] ; one lane writes 4 contiguous f32 ----
  {
    const int sl = lane >> 3, ec = (lane & 7) * 4;
    float w0 = __shfl(wts[0], sl * 4, 32);
    float wA = __shfl(wts[1], sl * 4, 32);
    float wB = __shfl(wts[2], sl * 4, 32);
    float wCv = __shfl(wts[3], sl * 4, 32);
    v4f x0 = *(const v4f*)(myX + (sl * 4 + 0) * 32 + ec);
    v4f x1 = *(const v4f*)(myX + (sl * 4 + 1) * 32 + ec);
    v4f x2 = *(const v4f*)(myX + (sl * 4 + 2) * 32 + ec);
    v4f x3 = *(const v4f*)(myX + (sl * 4 + 3) * 32 + ec);
    v4f f;
#pragma unroll
    for (int j = 0; j < 4; ++j)
      f[j] = x0[j] * w0 + x1[j] * wA + x2[j] * wB + x3[j] * wCv;
    *(v4f*)(out_fused + (size_t)(sampleBase + sl) * 32 + ec) = f;
  }
}

extern "C" void kernel_launch(void* const* d_in, const int* in_sizes, int n_in,
                              void* d_out, int out_size, void* d_ws, size_t ws_size,
                              hipStream_t stream) {
  (void)n_in; (void)out_size; (void)d_ws; (void)ws_size;
  const float* tokens     = (const float*)d_in[0];
  const float* amask      = (const float*)d_in[1];
  const float* type_emb   = (const float*)d_in[2];
  const float* in_proj_w  = (const float*)d_in[3];
  const float* in_proj_b  = (const float*)d_in[4];
  const float* out_proj_w = (const float*)d_in[5];
  const float* out_proj_b = (const float*)d_in[6];
  const float* ln1_s      = (const float*)d_in[7];
  const float* ln1_b      = (const float*)d_in[8];
  const float* w1         = (const float*)d_in[9];
  const float* b1         = (const float*)d_in[10];
  const float* w2         = (const float*)d_in[11];
  const float* b2         = (const float*)d_in[12];
  const float* ln2_s      = (const float*)d_in[13];
  const float* ln2_b      = (const float*)d_in[14];
  const float* gate_w     = (const float*)d_in[15];
  const float* gate_b     = (const float*)d_in[16];

  int Ntotal = in_sizes[0] / 128;               // tokens is [N,4,32] fp32
  float* out_fused   = (float*)d_out;           // [N,32]
  float* out_weights = out_fused + (size_t)Ntotal * 32;  // [N,4]

  int samplesPerBlock = WAVES_PER_BLOCK * SAMPLES_PER_WAVE; // 16
  int blocks = (Ntotal + samplesPerBlock - 1) / samplesPerBlock;
  tlf_fused_kernel<<<blocks, THREADS_PER_BLOCK, 0, stream>>>(
      tokens, amask, type_emb, in_proj_w, in_proj_b, out_proj_w, out_proj_b,
      ln1_s, ln1_b, w1, b1, w2, b2, ln2_s, ln2_b, gate_w, gate_b,
      out_fused, out_weights, Ntotal);
}
